// M3PlasticPolicy_87668872446433
// MI455X (gfx1250) — matrix operand order repaired and checked by reading, not compiled, not run on hardware
//
#include <hip/hip_runtime.h>
#include <hip/hip_bf16.h>
#include <math.h>

// ---------------------------------------------------------------------------
// M3PlasticPolicy forward for MI455X (gfx1250, wave32, WMMA, async-to-LDS).
// B=4, S=256, E=1024, H=2048, L=4, V=32000.
// ---------------------------------------------------------------------------

#define VOCAB   32000
#define EMBED   1024
#define HIDDEN  2048
#define NLAYERS 4
#define BATCH   4
#define SEQ     256
#define LN_EPS  1e-5f

// GEMM blocking: the 262MB W_out stream (HBM-resident) is read only
// M/BM = 4 times; the 8MB activation operand re-reads hit the 192MB L2.
#define BM 256
#define BN 128
#define GEMM_THREADS 512   // 16 waves; wave grid 4(M) x 4(N); 8 tiles/wave

typedef __attribute__((ext_vector_type(16))) __bf16 v16bf;
typedef __attribute__((ext_vector_type(8)))  float  v8f;

// ---------------------------------------------------------------------------
// CDNA5 async-to-LDS helpers (ASYNCcnt-tracked, 16B per lane per issue)
// ---------------------------------------------------------------------------
__device__ __forceinline__ void async_copy16(void* lds, const void* g) {
#if defined(__HIP_DEVICE_COMPILE__)
    asm volatile("global_load_async_to_lds_b128 %0, %1, off"
                 :: "v"((unsigned)(size_t)lds),
                    "v"((unsigned long long)(size_t)g)
                 : "memory");
#else
    __builtin_memcpy(lds, g, 16);
#endif
}
__device__ __forceinline__ void async_wait_all() {
#if defined(__HIP_DEVICE_COMPILE__)
    asm volatile("s_wait_asynccnt 0x0" ::: "memory");
#endif
}

__device__ __forceinline__ void cvt4(v16bf& dst, int base, float4 f) {
    dst[base + 0] = (__bf16)f.x;
    dst[base + 1] = (__bf16)f.y;
    dst[base + 2] = (__bf16)f.z;
    dst[base + 3] = (__bf16)f.w;
}

// ---------------------------------------------------------------------------
// Scalar reductions: abs-mean partials (fixed-order, deterministic)
// ---------------------------------------------------------------------------
__global__ __launch_bounds__(256) void k_abs_partial(const float* __restrict__ src,
                                                     int n, float* __restrict__ part) {
    __shared__ float sm[256];
    float s = 0.f;
    for (int i = blockIdx.x * 256 + threadIdx.x; i < n; i += 256 * 256)
        s += fabsf(src[i]);
    sm[threadIdx.x] = s;
    __syncthreads();
    for (int st = 128; st; st >>= 1) {
        if (threadIdx.x < st) sm[threadIdx.x] += sm[threadIdx.x + st];
        __syncthreads();
    }
    if (threadIdx.x == 0) part[blockIdx.x] = sm[0];
}

// scalars[0]=gating, scalars[1]=s_in, scalars[2..5]=s_l
__global__ void k_finalize(const float* __restrict__ part,
                           const float* __restrict__ affect,
                           float* __restrict__ scalars) {
    if (threadIdx.x != 0 || blockIdx.x != 0) return;
    const float cnt[5] = {(float)(HIDDEN * EMBED), (float)(HIDDEN * HIDDEN),
                          (float)(HIDDEN * HIDDEN), (float)(HIDDEN * HIDDEN),
                          (float)(HIDDEN * HIDDEN)};
    for (int t = 0; t < 5; ++t) {
        float s = 0.f;
        for (int i = 0; i < 256; ++i) s += part[t * 256 + i];
        scalars[1 + t] = s / cnt[t] + 1e-8f;
    }
    float ar = 0.f;
    for (int b = 0; b < BATCH; ++b) {
        float ss = 0.f;
        for (int j = 0; j < 16; ++j) { float v = affect[b * 16 + j]; ss += v * v; }
        ar += sqrtf(ss);
    }
    ar *= (1.0f / BATCH);
    float g = 1.0f + ar;                 // NEURO_SCALE*10 == 1.0
    scalars[0] = fminf(5.0f, fmaxf(0.5f, g));
}

__global__ void k_blsum(const float* __restrict__ bl, float* __restrict__ out) {
    int h = blockIdx.x * blockDim.x + threadIdx.x;
    if (h < HIDDEN)
        out[h] = bl[h] + bl[HIDDEN + h] + bl[2 * HIDDEN + h] + bl[3 * HIDDEN + h];
}

// ---------------------------------------------------------------------------
// Effective weights: ternary quant (absmean, RNE round) + gated plastic trace
// ---------------------------------------------------------------------------
__device__ __forceinline__ float qtern(float w, float s) {
    float q = rintf(w / s);
    return fminf(1.f, fmaxf(-1.f, q)) * s;
}

__global__ __launch_bounds__(256) void k_build_wineff(const float* __restrict__ W,
                                                      const float* __restrict__ tr,
                                                      const float* __restrict__ sc,
                                                      float* __restrict__ out, int n) {
    const float s = sc[1], g = sc[0];
    for (int i = blockIdx.x * blockDim.x + threadIdx.x; i < n;
         i += gridDim.x * blockDim.x)
        out[i] = qtern(W[i], s) + g * tr[i];
}

__global__ __launch_bounds__(256) void k_build_wsum(const float* __restrict__ Wl,
                                                    const float* __restrict__ tr,
                                                    const float* __restrict__ sc,
                                                    float* __restrict__ out, int n) {
    const float g = sc[0];
    for (int i = blockIdx.x * blockDim.x + threadIdx.x; i < n;
         i += gridDim.x * blockDim.x) {
        float a = 0.f;
        #pragma unroll
        for (int l = 0; l < NLAYERS; ++l) {
            float s = sc[2 + l];
            a += qtern(Wl[(size_t)l * n + i], s) + g * tr[(size_t)l * n + i];
        }
        out[i] = a;
    }
}

// ---------------------------------------------------------------------------
// Generic GEMM: C[M,N] = A[M,K] * Bt[N,K]^T + bias[N]
//   - double-buffered async-to-LDS staging: chunk k+1 streams into the other
//     LDS buffer while the matrix pipe consumes chunk k (1 barrier/chunk)
//   - fp32 -> bf16 in-register during fragment build (v_cvt_pk_bf16_f32)
//   - optional A row indirection (rowidx) fuses the embedding gather
// Block tile BM x BN, 16 waves, each wave owns 64(M) x 32(N) = 8 WMMA tiles.
// ---------------------------------------------------------------------------
__global__ __launch_bounds__(GEMM_THREADS) void k_gemm_bt(
        const float* __restrict__ A, const int* __restrict__ rowidx,
        const float* __restrict__ Bt, const float* __restrict__ bias,
        float* __restrict__ C, int M, int N, int K) {
    __shared__ __align__(16) float sA[2][BM][32];
    __shared__ __align__(16) float sB[2][BN][32];

    const int tid  = threadIdx.x;
    const int lane = tid & 31;
    const int wave = tid >> 5;
    const int row0 = blockIdx.y * BM;
    const int col0 = blockIdx.x * BN;
    const int wr   = wave >> 2;          // 0..3  -> M offset wr*64
    const int wc   = wave & 3;           // 0..3  -> N offset wc*32
    const int half = lane >> 4;          // WMMA K-half selector
    const int lm   = lane & 15;

    v8f acc[4][2];
    #pragma unroll
    for (int mi = 0; mi < 4; ++mi)
        #pragma unroll
        for (int ni = 0; ni < 2; ++ni)
            #pragma unroll
            for (int v = 0; v < 8; ++v) acc[mi][ni][v] = 0.f;

    auto stage = [&](int k0, int buf) {
        for (int i = tid; i < BM * 8; i += GEMM_THREADS) {
            int r = i >> 3, c = (i & 7) * 4;
            int gr = rowidx ? rowidx[row0 + r] : (row0 + r);
            async_copy16(&sA[buf][r][c], &A[(size_t)gr * K + k0 + c]);
        }
        for (int i = tid; i < BN * 8; i += GEMM_THREADS) {
            int r = i >> 3, c = (i & 7) * 4;
            async_copy16(&sB[buf][r][c], &Bt[(size_t)(col0 + r) * K + k0 + c]);
        }
    };

    const int NK = K / 32;
    stage(0, 0);
    async_wait_all();
    __syncthreads();

    for (int kc = 0; kc < NK; ++kc) {
        const int buf = kc & 1;
        if (kc + 2 < NK)   // L2 prefetch two chunks ahead on the weight stream
            __builtin_prefetch(
                &Bt[(size_t)(col0 + (tid & (BN - 1))) * K + (kc + 2) * 32], 0, 1);
        if (kc + 1 < NK)   // async-stage next chunk into the other buffer
            stage((kc + 1) * 32, buf ^ 1);

        // ---- compute on current buffer ----
        // B fragments: lane = N, K runs [8h..8h+7] and [16+8h..23+8h]; two aligned
        // float4 runs -> batched ds_load_b128 then packed cvt.
        v16bf bfr[2];
        #pragma unroll
        for (int ni = 0; ni < 2; ++ni) {
            const float4* bs = (const float4*)&sB[buf][wc * 32 + ni * 16 + lm][0];
            float4 b0 = bs[2 * half + 0], b1 = bs[2 * half + 1];
            float4 b2 = bs[2 * half + 4], b3 = bs[2 * half + 5];
            cvt4(bfr[ni], 0, b0);  cvt4(bfr[ni], 4, b1);
            cvt4(bfr[ni], 8, b2);  cvt4(bfr[ni], 12, b3);
        }
        // A fragments (K runs [8h..8h+7] and [16+8h..23+8h]) + 8 WMMAs.
        #pragma unroll
        for (int mi = 0; mi < 4; ++mi) {
            const float4* as = (const float4*)&sA[buf][wr * 64 + mi * 16 + lm][0];
            float4 a0 = as[2 * half + 0], a1 = as[2 * half + 1];
            float4 a2 = as[2 * half + 4], a3 = as[2 * half + 5];
            v16bf af;
            cvt4(af, 0, a0);  cvt4(af, 4, a1);
            cvt4(af, 8, a2);  cvt4(af, 12, a3);
            #pragma unroll
            for (int ni = 0; ni < 2; ++ni)
                acc[mi][ni] = __builtin_amdgcn_wmma_f32_16x16x32_bf16(
                    false, af, false, bfr[ni], (short)0, acc[mi][ni], false, false);
        }

        async_wait_all();    // next-chunk staging complete (own wave)
        __syncthreads();     // ... and published by all waves; LDS reads done
    }

    #pragma unroll
    for (int ni = 0; ni < 2; ++ni) {
        const int col = col0 + wc * 32 + ni * 16 + lm;
        const float bv = bias[col];
        #pragma unroll
        for (int mi = 0; mi < 4; ++mi)
            #pragma unroll
            for (int v = 0; v < 8; ++v) {
                int row = row0 + wr * 64 + mi * 16 + v + 8 * half;
                C[(size_t)row * N + col] = acc[mi][ni][v] + bv;
            }
    }
}

// ---------------------------------------------------------------------------
// Recurrent scan: 4 blocks (one per batch row), h in LDS, 256 serial steps.
// h_ff = h @ W_sum^T ; h_new = LN(gelu_exact(h + h_in_t + h_ff + b_l_sum)).
// W_sum (16 MB) lives in L2; lanes read float4 (k = 128j + 4*lane), which is
// contiguous 512B per wave -> global_load_b128 / ds_load_b128.
// ---------------------------------------------------------------------------
__global__ __launch_bounds__(1024) void k_scan(const float* __restrict__ h0,
                                               const float* __restrict__ h_in,
                                               const float* __restrict__ Wsum,
                                               const float* __restrict__ blsum,
                                               const float* __restrict__ gam,
                                               const float* __restrict__ bet,
                                               float* __restrict__ outs,
                                               float* __restrict__ hfinal) {
    __shared__ __align__(16) float hc[HIDDEN];
    __shared__ __align__(16) float pre[HIDDEN];
    __shared__ float redS[32], redQ[32];

    const int b    = blockIdx.x;
    const int tid  = threadIdx.x;
    const int lane = tid & 31;
    const int wave = tid >> 5;

    for (int g = tid; g < HIDDEN; g += 1024) hc[g] = h0[b * HIDDEN + g];
    __syncthreads();

    for (int t = 0; t < SEQ; ++t) {
        const float* hrow = h_in + (size_t)(b * SEQ + t) * HIDDEN;

        // Each of 32 waves computes 64 outputs; lanes split K in float4s.
        for (int o = 0; o < 64; ++o) {
            int g = wave * 64 + o;
            const float4* w4 = (const float4*)(Wsum + (size_t)g * HIDDEN);
            const float4* h4 = (const float4*)hc;
            float p = 0.f;
            #pragma unroll 4
            for (int j = 0; j < HIDDEN / 128; ++j) {
                float4 hv = h4[j * 32 + lane];
                float4 wv = w4[j * 32 + lane];
                p = fmaf(hv.x, wv.x, p);
                p = fmaf(hv.y, wv.y, p);
                p = fmaf(hv.z, wv.z, p);
                p = fmaf(hv.w, wv.w, p);
            }
            for (int off = 16; off; off >>= 1) p += __shfl_xor(p, off, 32);
            if (lane == 0) {
                float x = hc[g] + hrow[g] + p + blsum[g];
                pre[g] = 0.5f * x * (1.0f + erff(x * 0.70710678118654752f));
            }
        }
        __syncthreads();

        // Block-wide mean / mean-of-squares over 2048 values.
        float v0 = pre[tid], v1 = pre[tid + 1024];
        float s = v0 + v1;
        float q = v0 * v0 + v1 * v1;
        for (int off = 16; off; off >>= 1) {
            s += __shfl_xor(s, off, 32);
            q += __shfl_xor(q, off, 32);
        }
        if (lane == 0) { redS[wave] = s; redQ[wave] = q; }
        __syncthreads();
        if (wave == 0) {
            float s2 = redS[lane], q2 = redQ[lane];
            for (int off = 16; off; off >>= 1) {
                s2 += __shfl_xor(s2, off, 32);
                q2 += __shfl_xor(q2, off, 32);
            }
            if (lane == 0) {
                redS[0] = s2 * (1.f / HIDDEN);
                redQ[0] = q2 * (1.f / HIDDEN);
            }
        }
        __syncthreads();

        const float mu   = redS[0];
        const float var  = redQ[0] - mu * mu;
        const float rstd = rsqrtf(var + LN_EPS);

        float* orow = outs + (size_t)(b * SEQ + t) * HIDDEN;
        for (int g = tid; g < HIDDEN; g += 1024) {
            float hn = (pre[g] - mu) * rstd * gam[g] + bet[g];
            orow[g] = hn;
            hc[g]   = hn;
        }
        __syncthreads();
    }

    for (int g = tid; g < HIDDEN; g += 1024) hfinal[b * HIDDEN + g] = hc[g];
}

// ---------------------------------------------------------------------------
// Driver
// ---------------------------------------------------------------------------
extern "C" void kernel_launch(void* const* d_in, const int* in_sizes, int n_in,
                              void* d_out, int out_size, void* d_ws, size_t ws_size,
                              hipStream_t stream) {
    const int*   input_ids    = (const int*)  d_in[0];
    const float* hidden0      = (const float*)d_in[1];
    const float* affect       = (const float*)d_in[2];
    const float* emb          = (const float*)d_in[3];
    const float* W_in         = (const float*)d_in[4];
    const float* b_in         = (const float*)d_in[5];
    const float* trace_in     = (const float*)d_in[6];
    const float* W_layers     = (const float*)d_in[7];
    const float* b_layers     = (const float*)d_in[8];
    const float* trace_layers = (const float*)d_in[9];
    const float* W_out        = (const float*)d_in[10];
    const float* b_out        = (const float*)d_in[11];
    const float* ln_g         = (const float*)d_in[12];
    const float* ln_b         = (const float*)d_in[13];

    float* ws      = (float*)d_ws;
    float* scalars = ws;                                       // 16 floats
    float* part    = ws + 16;                                  // 5*256
    float* blsum   = ws + 2048;                                // 2048
    float* wineff  = ws + 4096;                                // 2048*1024
    float* wsum    = wineff + (size_t)HIDDEN * EMBED;          // 2048*2048
    float* hin     = wsum + (size_t)HIDDEN * HIDDEN;           // 1024*2048
    float* outsbuf = hin + (size_t)BATCH * SEQ * HIDDEN;       // 1024*2048

    float* logits  = (float*)d_out;                            // [4,256,32000]
    float* hfinal  = logits + (size_t)BATCH * SEQ * VOCAB;     // [4,2048]

    const int nWin = HIDDEN * EMBED;     // 2,097,152
    const int nWl  = HIDDEN * HIDDEN;    // 4,194,304

    // 1) Quant scale reductions (fixed-order, deterministic).
    k_abs_partial<<<256, 256, 0, stream>>>(W_in, nWin, part);
    for (int l = 0; l < NLAYERS; ++l)
        k_abs_partial<<<256, 256, 0, stream>>>(W_layers + (size_t)l * nWl, nWl,
                                               part + (1 + l) * 256);
    k_finalize<<<1, 64, 0, stream>>>(part, affect, scalars);

    // 2) Bias sum and effective weights (quant + gated plastic traces).
    k_blsum<<<(HIDDEN + 255) / 256, 256, 0, stream>>>(b_layers, blsum);
    k_build_wineff<<<2048, 256, 0, stream>>>(W_in, trace_in, scalars, wineff, nWin);
    k_build_wsum<<<4096, 256, 0, stream>>>(W_layers, trace_layers, scalars, wsum, nWl);

    // 3) Input projection GEMM (WMMA bf16) with fused embedding gather:
    //    hin[m,:] = emb[input_ids[m],:] @ Win_eff^T + b_in.
    {
        dim3 grid(HIDDEN / BN, (BATCH * SEQ) / BM);            // (16, 4)
        k_gemm_bt<<<grid, GEMM_THREADS, 0, stream>>>(
            emb, input_ids, wineff, b_in, hin, BATCH * SEQ, HIDDEN, EMBED);
    }

    // 4) Serial recurrence (4 independent blocks, one per batch row).
    k_scan<<<BATCH, 1024, 0, stream>>>(hidden0, hin, wsum, blsum, ln_g, ln_b,
                                       outsbuf, hfinal);

    // 5) Logits GEMM (WMMA bf16): logits = outputs @ W_out^T + b_out.
    {
        dim3 grid(VOCAB / BN, (BATCH * SEQ) / BM);             // (250, 4)
        k_gemm_bt<<<grid, GEMM_THREADS, 0, stream>>>(
            outsbuf, (const int*)nullptr, W_out, b_out, logits,
            BATCH * SEQ, VOCAB, HIDDEN);
    }
}
